// EvoVGM_GTR_70489003262335
// MI455X (gfx1250) — matrix operand order repaired and checked by reading, not compiled, not run on hardware
//
#include <hip/hip_runtime.h>
#include <hip/hip_bf16.h>
#include <stdint.h>

// ---------------------------------------------------------------------------
// CDNA5 (gfx1250) EvoVGM-GTR step.
//   K0 conv     : fp32 weights -> bf16, transposed to [N][K] for WMMA B-frags
//   K1 small    : b_ws / Dirichlet samples / KLs / 4x4 expm -> P[S,128,4,4]
//   K2 mlp      : fused 3-layer MLP with v_wmma_f32_16x16x32_bf16, log_softmax
//   K3 decoder  : P DMA'd into 80KB LDS via TDM tensor_load_to_lds (TENSORcnt),
//                 gumbel-softmax, per-(site,branch) 4x4 matvec + log-lik
//   K4 finalize : reduce logp_m, assemble elbo / logp / kl_total
// ---------------------------------------------------------------------------

typedef __attribute__((ext_vector_type(16))) __bf16    v16bf;
typedef __attribute__((ext_vector_type(8)))  float     v8f;
typedef __attribute__((ext_vector_type(4)))  unsigned  u32x4;
typedef __attribute__((ext_vector_type(8)))  unsigned  u32x8;

#define NS_MAX    10000
#define M_BR      128
#define S_SAMP    10
#define EPSV      1e-10f
#define INV_TEMP  10.0f

// ---- workspace layout (float offsets) ----
#define P_OFF      0        // 20480 : P[S][128][4][4]
#define BWS_OFF    20480    // 1280
#define RWS_OFF    21760    // 60
#define FWS_OFF    21820    // 40
#define KLABRF_OFF 21860    // 1
#define AKL_OFF    21861    // 1
#define LOGPM_OFF  21862    // 128
#define LOGPI_OFF  22016    // 40000
#define W1T_BYTE   248064   // 64*512 bf16  (32B aligned)
#define W2T_BYTE   313600   // 64*64  bf16
#define W3T_BYTE   321792   // 16*64  bf16 (zero-padded cols 4..15)

// ---- d_out layout (float offsets) ----
#define O_ELBO 0
#define O_LOGP 1
#define O_KL   2
#define O_ANC  3
#define O_BR   40003
#define O_RT   40131
#define O_FQ   40137
#define O_XR   40141

// ------------------------- RNG helpers -------------------------------------
__device__ __forceinline__ unsigned mixu(unsigned a, unsigned b) {
  unsigned x = a * 0x9E3779B1u + b * 0x85EBCA77u + 0x165667B1u;
  x ^= x >> 16; x *= 0x7FEB352Du;
  x ^= x >> 15; x *= 0x846CA68Bu;
  x ^= x >> 16;
  return x;
}
__device__ __forceinline__ float u01(unsigned h) {
  return ((float)(h >> 8) + 0.5f) * (1.0f / 16777216.0f);
}
__device__ __forceinline__ float nrm01(unsigned a, unsigned b) {
  float u1 = u01(mixu(a, b));
  float u2 = u01(mixu(a, b ^ 0xDEADBEEFu));
  return sqrtf(-2.0f * logf(u1)) * cosf(6.28318530718f * u2);
}
__device__ float digamma_f(float x) {
  float r = 0.0f;
  while (x < 6.0f) { r -= 1.0f / x; x += 1.0f; }
  float inv = 1.0f / x, inv2 = inv * inv;
  return r + logf(x) - 0.5f * inv -
         inv2 * (1.0f / 12.0f - inv2 * (1.0f / 120.0f - inv2 * (1.0f / 252.0f)));
}
__device__ float gamma_sample(float alpha, unsigned key) {
  float boost = 1.0f;
  unsigned c = 0;
  if (alpha < 1.0f) {
    boost = powf(u01(mixu(key, 17u)), 1.0f / alpha);
    alpha += 1.0f;
  }
  float d = alpha - (1.0f / 3.0f);
  float cc = 1.0f / sqrtf(9.0f * d);
  for (int it = 0; it < 16; ++it) {
    float z = nrm01(key, 100u + c); c++;
    float v = 1.0f + cc * z;
    if (v <= 0.0f) continue;
    v = v * v * v;
    float u = u01(mixu(key, 300u + c)); c++;
    if (logf(u) < 0.5f * z * z + d - d * v + d * logf(v))
      return boost * d * v;
  }
  return boost * d;
}
__device__ float dirichlet_kl_vs_ones(const float* logalpha, int K) {
  float a0 = 0.0f;
  for (int i = 0; i < K; ++i) a0 += expf(logalpha[i]);
  float kl = lgammaf(a0) - lgammaf((float)K);
  float dg0 = digamma_f(a0);
  for (int i = 0; i < K; ++i) {
    float ai = expf(logalpha[i]);
    kl -= lgammaf(ai);
    kl += (ai - 1.0f) * (digamma_f(ai) - dg0);
  }
  return kl;
}

// ------------------------- 4x4 expm ----------------------------------------
__device__ __forceinline__ void mm4(const float* A, const float* B, float* C) {
#pragma unroll
  for (int a = 0; a < 4; ++a)
#pragma unroll
    for (int b = 0; b < 4; ++b) {
      float acc = 0.0f;
#pragma unroll
      for (int k = 0; k < 4; ++k) acc += A[a * 4 + k] * B[k * 4 + b];
      C[a * 4 + b] = acc;
    }
}
__device__ void expm4(const float* M, float* O) {
  float nrm = 0.0f;
#pragma unroll
  for (int a = 0; a < 4; ++a) {
    float rs = 0.0f;
#pragma unroll
    for (int b = 0; b < 4; ++b) rs += fabsf(M[a * 4 + b]);
    nrm = fmaxf(nrm, rs);
  }
  int j = 0; float sc = 1.0f;
  while (nrm > 0.5f && j < 30) { nrm *= 0.5f; sc *= 0.5f; ++j; }
  float A[16], S[16], T[16];
#pragma unroll
  for (int e = 0; e < 16; ++e) {
    A[e] = M[e] * sc;
    float id = ((e & 3) == (e >> 2)) ? 1.0f : 0.0f;
    S[e] = id; T[e] = id;
  }
  for (int k = 1; k <= 10; ++k) {
    float Tn[16]; mm4(T, A, Tn);
    float inv = 1.0f / (float)k;
#pragma unroll
    for (int e = 0; e < 16; ++e) { T[e] = Tn[e] * inv; S[e] += T[e]; }
  }
  for (int q = 0; q < j; ++q) {
    float Sn[16]; mm4(S, S, Sn);
#pragma unroll
    for (int e = 0; e < 16; ++e) S[e] = Sn[e];
  }
#pragma unroll
  for (int e = 0; e < 16; ++e) O[e] = S[e];
}

// ------------------------- K0: weight conversion ---------------------------
__global__ __launch_bounds__(256) void conv_kernel(
    const float* __restrict__ W1, const float* __restrict__ W2,
    const float* __restrict__ W3, __bf16* __restrict__ w1t,
    __bf16* __restrict__ w2t, __bf16* __restrict__ w3t) {
  int tid = threadIdx.x;
  for (int i = tid; i < 64 * 512; i += 256) {
    int n = i >> 9, k = i & 511;
    w1t[i] = (__bf16)W1[k * 64 + n];
  }
  for (int i = tid; i < 64 * 64; i += 256) {
    int n = i >> 6, k = i & 63;
    w2t[i] = (__bf16)W2[k * 64 + n];
  }
  for (int i = tid; i < 16 * 64; i += 256) {
    int n = i >> 6, k = i & 63;
    w3t[i] = (n < 4) ? (__bf16)W3[k * 4 + n] : (__bf16)0.0f;
  }
}

// ------------------------- K1: small head (b, r, f, expm) ------------------
__global__ __launch_bounds__(256) void small_kernel(
    const float* __restrict__ site_counts, const float* __restrict__ b_mu,
    const float* __restrict__ b_logsig, const float* __restrict__ r_la,
    const float* __restrict__ f_la, float* __restrict__ wsf,
    float* __restrict__ out, int n_sites) {
  float* P    = wsf + P_OFF;
  float* b_ws = wsf + BWS_OFF;
  float* r_ws = wsf + RWS_OFF;
  float* f_ws = wsf + FWS_OFF;
  int tid = threadIdx.x;

  __shared__ float red[256];
  __shared__ float g_r[60], g_f[40], rs_r[10], rs_f[10];
  __shared__ float sh_bkl, sh_N, sh_rkl, sh_fkl;

  // zero global accumulators (fresh every call)
  if (tid == 0) wsf[AKL_OFF] = 0.0f;
  if (tid < 128) wsf[LOGPM_OFF + tid] = 0.0f;

  // --- b_ws ~ LogNormal ---
  for (int idx = tid; idx < S_SAMP * M_BR; idx += 256) {
    int m = idx & 127;
    float sig = expf(b_logsig[m]);
    float z = nrm01(0x1234u + idx, 0xB00B5u);
    b_ws[idx] = expf(b_mu[m] + sig * z);
  }
  __syncthreads();

  // branches mean + per-m KL
  float bkl = 0.0f;
  if (tid < 128) {
    float acc = 0.0f;
#pragma unroll
    for (int s = 0; s < S_SAMP; ++s) acc += b_ws[s * M_BR + tid];
    out[O_BR + tid] = acc * (1.0f / (float)S_SAMP);
    float sig = expf(b_logsig[tid]);
    float mu0 = logf(0.1f), s0 = 0.1f;
    float dmu = b_mu[tid] - mu0;
    bkl = logf(s0 / sig) + (sig * sig + dmu * dmu) / (2.0f * s0 * s0) - 0.5f;
  }
  red[tid] = bkl; __syncthreads();
  for (int off = 128; off; off >>= 1) {
    if (tid < off) red[tid] += red[tid + off];
    __syncthreads();
  }
  if (tid == 0) sh_bkl = red[0];
  __syncthreads();

  // N = sum(site_counts)
  float np = 0.0f;
  for (int i = tid; i < n_sites; i += 256) np += site_counts[i];
  red[tid] = np; __syncthreads();
  for (int off = 128; off; off >>= 1) {
    if (tid < off) red[tid] += red[tid + off];
    __syncthreads();
  }
  if (tid == 0) sh_N = red[0];

  // --- Dirichlet gammas ---
  if (tid < 60) g_r[tid] = gamma_sample(expf(r_la[tid % 6]), mixu(0x6001u, tid));
  if (tid >= 64 && tid < 104) {
    int t = tid - 64;
    g_f[t] = gamma_sample(expf(f_la[t % 4]), mixu(0x7001u, t));
  }
  __syncthreads();
  if (tid < 10) {
    float a = 0.0f, b = 0.0f;
#pragma unroll
    for (int k = 0; k < 6; ++k) a += g_r[tid * 6 + k];
#pragma unroll
    for (int k = 0; k < 4; ++k) b += g_f[tid * 4 + k];
    rs_r[tid] = a; rs_f[tid] = b;
  }
  __syncthreads();
  if (tid < 60) r_ws[tid] = g_r[tid] / rs_r[tid / 6];
  if (tid < 40) f_ws[tid] = g_f[tid] / rs_f[tid / 4];
  __syncthreads();

  if (tid < 6) {
    float a = 0.0f;
#pragma unroll
    for (int s = 0; s < S_SAMP; ++s) a += r_ws[s * 6 + tid];
    out[O_RT + tid] = a * (1.0f / (float)S_SAMP);
  }
  if (tid < 4) {
    float a = 0.0f;
#pragma unroll
    for (int s = 0; s < S_SAMP; ++s) a += f_ws[s * 4 + tid];
    out[O_FQ + tid] = a * (1.0f / (float)S_SAMP);
  }
  if (tid == 0) {
    float rla[6], fla[4];
    for (int i = 0; i < 6; ++i) rla[i] = r_la[i];
    for (int i = 0; i < 4; ++i) fla[i] = f_la[i];
    sh_rkl = dirichlet_kl_vs_ones(rla, 6);
    sh_fkl = dirichlet_kl_vs_ones(fla, 4);
  }
  __syncthreads();
  if (tid == 0) wsf[KLABRF_OFF] = sh_N * (sh_bkl + sh_rkl + sh_fkl);

  // --- Q build + expm per (s, m) ---
  const int iu[6] = {0, 0, 0, 1, 1, 2};
  const int ju[6] = {1, 2, 3, 2, 3, 3};
  for (int idx = tid; idx < S_SAMP * M_BR; idx += 256) {
    int s = idx >> 7;
    float r[6], f[4];
#pragma unroll
    for (int e = 0; e < 6; ++e) r[e] = r_ws[s * 6 + e];
#pragma unroll
    for (int e = 0; e < 4; ++e) f[e] = f_ws[s * 4 + e];
    float R[16];
#pragma unroll
    for (int e = 0; e < 16; ++e) R[e] = 0.0f;
#pragma unroll
    for (int e = 0; e < 6; ++e) {
      R[iu[e] * 4 + ju[e]] = r[e];
      R[ju[e] * 4 + iu[e]] = r[e];
    }
    float Q[16];
#pragma unroll
    for (int a = 0; a < 4; ++a) {
      float rs = 0.0f;
#pragma unroll
      for (int b = 0; b < 4; ++b) { Q[a * 4 + b] = R[a * 4 + b] * f[b]; rs += Q[a * 4 + b]; }
      Q[a * 4 + a] = -rs;
    }
    float norm = 0.0f;
#pragma unroll
    for (int a = 0; a < 4; ++a) norm -= f[a] * Q[a * 4 + a];
    float scale = b_ws[idx] / norm;
    float Mm[16];
#pragma unroll
    for (int e = 0; e < 16; ++e) Mm[e] = Q[e] * scale;
    float Pe[16];
    expm4(Mm, Pe);
#pragma unroll
    for (int e = 0; e < 16; ++e) P[idx * 16 + e] = Pe[e];
  }
}

// ------------------------- K2: fused WMMA MLP ------------------------------
// block = 256 threads = 8 waves; wave w handles rows [blk*128 + 16w, +16)
__global__ __launch_bounds__(256) void mlp_kernel(
    const float* __restrict__ sites, const float* __restrict__ site_counts,
    const float* __restrict__ b1, const float* __restrict__ b2,
    const float* __restrict__ b3, const __bf16* __restrict__ w1t,
    const __bf16* __restrict__ w2t, const __bf16* __restrict__ w3t,
    float* __restrict__ logpi_out, float* __restrict__ a_kl, int n_sites) {
  __shared__ __bf16 hs1[8][16][64];
  __shared__ __bf16 hs2[8][16][64];
  __shared__ float  lg[8][16][16];

  const int tid = threadIdx.x;
  const int w = tid >> 5, lane = tid & 31;
  const int r = lane & 15, kh = lane >> 4;   // A-frag row / K-half, B-frag col
  const int rowBase = blockIdx.x * 128 + w * 16;
  const int rr = min(rowBase + r, n_sites - 1);
  const float* xrow = sites + (size_t)rr * 512;

  // ---- GEMM1: X[16x512] @ W1[512x64] + b1, relu ----
  v8f acc[4];
#pragma unroll
  for (int nt = 0; nt < 4; ++nt)
#pragma unroll
    for (int v = 0; v < 8; ++v) acc[nt][v] = 0.0f;

  for (int k0 = 0; k0 < 512; k0 += 32) {
    v16bf a;
#pragma unroll
    for (int g = 0; g < 2; ++g) {
      const float* p = xrow + k0 + kh * 16 + g * 8;
#pragma unroll
      for (int i = 0; i < 8; ++i) a[g * 8 + i] = (__bf16)p[i];
    }
#pragma unroll
    for (int nt = 0; nt < 4; ++nt) {
      v16bf b = *(const v16bf*)(w1t + (size_t)(nt * 16 + r) * 512 + k0 + kh * 16);
      acc[nt] = __builtin_amdgcn_wmma_f32_16x16x32_bf16(
          false, a, false, b, (short)0, acc[nt], false, false);
    }
  }
#pragma unroll
  for (int nt = 0; nt < 4; ++nt) {
    float bias = b1[nt * 16 + r];
#pragma unroll
    for (int v = 0; v < 8; ++v)
      hs1[w][kh * 8 + v][nt * 16 + r] = (__bf16)fmaxf(acc[nt][v] + bias, 0.0f);
  }
  __syncthreads();

  // ---- GEMM2: h1[16x64] @ W2[64x64] + b2, relu ----
  v8f acc2[4];
#pragma unroll
  for (int nt = 0; nt < 4; ++nt)
#pragma unroll
    for (int v = 0; v < 8; ++v) acc2[nt][v] = 0.0f;
#pragma unroll
  for (int ks = 0; ks < 2; ++ks) {
    int k0 = ks * 32;
    v16bf a;
#pragma unroll
    for (int g = 0; g < 2; ++g) {
      const __bf16* p = &hs1[w][r][k0 + kh * 16 + g * 8];
#pragma unroll
      for (int i = 0; i < 8; ++i) a[g * 8 + i] = p[i];
    }
#pragma unroll
    for (int nt = 0; nt < 4; ++nt) {
      v16bf b = *(const v16bf*)(w2t + (size_t)(nt * 16 + r) * 64 + k0 + kh * 16);
      acc2[nt] = __builtin_amdgcn_wmma_f32_16x16x32_bf16(
          false, a, false, b, (short)0, acc2[nt], false, false);
    }
  }
#pragma unroll
  for (int nt = 0; nt < 4; ++nt) {
    float bias = b2[nt * 16 + r];
#pragma unroll
    for (int v = 0; v < 8; ++v)
      hs2[w][kh * 8 + v][nt * 16 + r] = (__bf16)fmaxf(acc2[nt][v] + bias, 0.0f);
  }
  __syncthreads();

  // ---- GEMM3: h2[16x64] @ W3[64x16(pad)] ----
  v8f acc3;
#pragma unroll
  for (int v = 0; v < 8; ++v) acc3[v] = 0.0f;
#pragma unroll
  for (int ks = 0; ks < 2; ++ks) {
    int k0 = ks * 32;
    v16bf a;
#pragma unroll
    for (int g = 0; g < 2; ++g) {
      const __bf16* p = &hs2[w][r][k0 + kh * 16 + g * 8];
#pragma unroll
      for (int i = 0; i < 8; ++i) a[g * 8 + i] = p[i];
    }
    v16bf b = *(const v16bf*)(w3t + (size_t)r * 64 + k0 + kh * 16);
    acc3 = __builtin_amdgcn_wmma_f32_16x16x32_bf16(
        false, a, false, b, (short)0, acc3, false, false);
  }
  float b3c = (r < 4) ? b3[r] : 0.0f;
#pragma unroll
  for (int v = 0; v < 8; ++v) lg[w][kh * 8 + v][r] = acc3[v] + b3c;
  __syncthreads();

  // ---- per-row log_softmax + a_kl ----
  if (lane < 16) {
    int row = rowBase + lane;
    if (row < n_sites) {
      float x0 = lg[w][lane][0], x1 = lg[w][lane][1];
      float x2 = lg[w][lane][2], x3 = lg[w][lane][3];
      float mx = fmaxf(fmaxf(x0, x1), fmaxf(x2, x3));
      float se = expf(x0 - mx) + expf(x1 - mx) + expf(x2 - mx) + expf(x3 - mx);
      float lse = mx + logf(se);
      float akl = 0.0f;
      float xs[4] = {x0, x1, x2, x3};
#pragma unroll
      for (int j = 0; j < 4; ++j) {
        float lp = xs[j] - lse;
        logpi_out[(size_t)row * 4 + j] = lp;
        float pi = expf(lp);
        akl += pi * (lp + 1.3862943611f);   // lp - log(0.25)
      }
      atomicAdd(a_kl, akl * site_counts[row]);
    }
  }
}

// ------------------------- K3: decoder -------------------------------------
// block = 128 threads (one per branch m); P[S,128,4,4] (80 KB) is DMA'd into
// LDS by the Tensor Data Mover (2-D D#: 4096 x 5 tile of 4-byte elements),
// tracked with TENSORcnt. CDNA5 allows 320 KB LDS per workgroup.
__global__ __launch_bounds__(128) void decoder_kernel(
    const float* __restrict__ sites, const float* __restrict__ site_counts,
    const float* __restrict__ P, const float* __restrict__ logpi,
    float* __restrict__ anc_out, float* __restrict__ xrec_out,
    float* __restrict__ logp_m, int n_sites) {
  __shared__ float Pl[S_SAMP * M_BR * 16];   // 80 KB
  __shared__ float al[S_SAMP][4];

  const int tid = threadIdx.x;   // == m

  // ---- TDM: tensor_load_to_lds (wave 0 issues; EXEC ignored by TDM) ----
  if (tid < 32) {
    unsigned lds_addr = (unsigned)(uintptr_t)(void*)&Pl[0];   // LDS byte offset
    unsigned long long ga = (unsigned long long)(uintptr_t)(const void*)P;
    // D# group 0: count=1 | lds_addr | global_addr[56:0] | type=2
    u32x4 g0;
    g0[0] = 1u;                                    // count=1, user desc
    g0[1] = lds_addr;                              // lds_addr
    g0[2] = (unsigned)ga;                          // global_addr[31:0]
    g0[3] = (unsigned)(ga >> 32) | (2u << 30);     // global_addr[56:32] | type=2
    // D# group 1: data_size=4B; tensor 4096x5; tile 4096x5; stride0=4096
    u32x8 g1;
    g1[0] = 2u << 16;          // workgroup_mask=0, data_size=2 (4 bytes)
    g1[1] = 4096u << 16;       // tensor_dim0[15:0] in [31:16]
    g1[2] = 5u << 16;          // tensor_dim0[31:16]=0, tensor_dim1[15:0]=5
    g1[3] = 4096u << 16;       // tensor_dim1[31:16]=0, tile_dim0=4096
    g1[4] = 5u;                // tile_dim1=5, tile_dim2=0
    g1[5] = 4096u;             // tensor_dim0_stride[31:0]
    g1[6] = 0u;                // stride0[47:32]=0, stride1[15:0]=0
    g1[7] = 0u;                // stride1[47:16]=0
    asm volatile("tensor_load_to_lds %0, %1"
                 :: "s"(g0), "s"(g1)
                 : "memory");
    __builtin_amdgcn_s_wait_tensorcnt(0);
  }
  __syncthreads();

  const int nStart = blockIdx.x * 16;
  const int nEnd = min(nStart + 16, n_sites);
  float logp_loc = 0.0f;

  for (int n = nStart; n < nEnd; ++n) {
    // prefetch next site row while we work on this one
    if (n + 1 < nEnd)
      __builtin_prefetch(sites + ((size_t)(n + 1) * M_BR + tid) * 4, 0, 1);

    // gumbel-softmax samples a_ws[s, n, :] (threads 0..9)
    if (tid < S_SAMP) {
      int s = tid;
      float x[4]; float mx = -1e30f;
#pragma unroll
      for (int j = 0; j < 4; ++j) {
        float lp = logpi[(size_t)n * 4 + j];
        float u = u01(mixu((unsigned)(n * 4 + j), 0xA5A50000u + (unsigned)s));
        float g = -logf(-logf(u + EPSV) + EPSV);
        x[j] = (lp + g) * INV_TEMP;
        mx = fmaxf(mx, x[j]);
      }
      float se = 0.0f;
#pragma unroll
      for (int j = 0; j < 4; ++j) { x[j] = expf(x[j] - mx); se += x[j]; }
      float inv = 1.0f / se;
#pragma unroll
      for (int j = 0; j < 4; ++j) al[s][j] = x[j] * inv;
    }
    __syncthreads();

    if (tid < 4) {
      float v = 0.0f;
#pragma unroll
      for (int s = 0; s < S_SAMP; ++s) v += al[s][tid];
      anc_out[(size_t)n * 4 + tid] = v * (1.0f / (float)S_SAMP);
    }

    const float* sv = sites + ((size_t)n * M_BR + tid) * 4;
    float s0 = sv[0], s1 = sv[1], s2 = sv[2], s3 = sv[3];
    float xr0 = 0, xr1 = 0, xr2 = 0, xr3 = 0, accll = 0.0f;
#pragma unroll
    for (int s = 0; s < S_SAMP; ++s) {
      const float* Pm = &Pl[(s * M_BR + tid) * 16];
      float a0 = al[s][0], a1 = al[s][1], a2 = al[s][2], a3 = al[s][3];
      float r0 = a0 * Pm[0] + a1 * Pm[4] + a2 * Pm[8]  + a3 * Pm[12];
      float r1 = a0 * Pm[1] + a1 * Pm[5] + a2 * Pm[9]  + a3 * Pm[13];
      float r2 = a0 * Pm[2] + a1 * Pm[6] + a2 * Pm[10] + a3 * Pm[14];
      float r3 = a0 * Pm[3] + a1 * Pm[7] + a2 * Pm[11] + a3 * Pm[15];
      xr0 += r0; xr1 += r1; xr2 += r2; xr3 += r3;
      float dot = r0 * s0 + r1 * s1 + r2 * s2 + r3 * s3;
      accll += logf(dot + EPSV);
    }
    float* xo = xrec_out + ((size_t)n * M_BR + tid) * 4;
    const float invS = 1.0f / (float)S_SAMP;
    xo[0] = xr0 * invS; xo[1] = xr1 * invS;
    xo[2] = xr2 * invS; xo[3] = xr3 * invS;
    logp_loc += accll * invS * site_counts[n];
    __syncthreads();   // al reused next iteration
  }
  atomicAdd(&logp_m[tid], logp_loc);
}

// ------------------------- K4: finalize ------------------------------------
__global__ __launch_bounds__(128) void finalize_kernel(
    const float* __restrict__ wsf, float* __restrict__ out) {
  __shared__ float red[128];
  int tid = threadIdx.x;
  red[tid] = wsf[LOGPM_OFF + tid];
  __syncthreads();
  for (int off = 64; off; off >>= 1) {
    if (tid < off) red[tid] += red[tid + off];
    __syncthreads();
  }
  if (tid == 0) {
    float lpt = red[0];
    float kl = wsf[KLABRF_OFF] + wsf[AKL_OFF];
    out[O_LOGP] = lpt;
    out[O_KL] = kl;
    out[O_ELBO] = lpt - (float)M_BR * 0.001f * kl;   // (logp - a*kl).sum over m
  }
}

// ------------------------- launch ------------------------------------------
extern "C" void kernel_launch(void* const* d_in, const int* in_sizes, int n_in,
                              void* d_out, int out_size, void* d_ws, size_t ws_size,
                              hipStream_t stream) {
  const float* sites       = (const float*)d_in[0];
  const float* site_counts = (const float*)d_in[1];
  const float* W1 = (const float*)d_in[2];
  const float* b1 = (const float*)d_in[3];
  const float* W2 = (const float*)d_in[4];
  const float* b2 = (const float*)d_in[5];
  const float* W3 = (const float*)d_in[6];
  const float* b3 = (const float*)d_in[7];
  const float* b_mu     = (const float*)d_in[8];
  const float* b_logsig = (const float*)d_in[9];
  const float* r_la     = (const float*)d_in[10];
  const float* f_la     = (const float*)d_in[11];

  float* out = (float*)d_out;
  float* wsf = (float*)d_ws;
  __bf16* w1t = (__bf16*)((char*)d_ws + W1T_BYTE);
  __bf16* w2t = (__bf16*)((char*)d_ws + W2T_BYTE);
  __bf16* w3t = (__bf16*)((char*)d_ws + W3T_BYTE);

  const int n_sites = in_sizes[0] / (M_BR * 4);

  hipLaunchKernelGGL(conv_kernel, dim3(1), dim3(256), 0, stream,
                     W1, W2, W3, w1t, w2t, w3t);
  hipLaunchKernelGGL(small_kernel, dim3(1), dim3(256), 0, stream,
                     site_counts, b_mu, b_logsig, r_la, f_la, wsf, out, n_sites);
  hipLaunchKernelGGL(mlp_kernel, dim3((n_sites + 127) / 128), dim3(256), 0, stream,
                     sites, site_counts, b1, b2, b3, w1t, w2t, w3t,
                     wsf + LOGPI_OFF, wsf + AKL_OFF, n_sites);
  hipLaunchKernelGGL(decoder_kernel, dim3((n_sites + 15) / 16), dim3(128), 0, stream,
                     sites, site_counts, wsf + P_OFF, wsf + LOGPI_OFF,
                     out + O_ANC, out + O_XR, wsf + LOGPM_OFF, n_sites);
  hipLaunchKernelGGL(finalize_kernel, dim3(1), dim3(128), 0, stream, wsf, out);
}
